// SwinStage0_23313082483333
// MI455X (gfx1250) — matrix operand order repaired and checked by reading, not compiled, not run on hardware
//
#include <hip/hip_runtime.h>
#include <hip/hip_bf16.h>
#include <math.h>

// ---------------------------------------------------------------------------
// CDNA5 (gfx1250) Swin stage0+merge+stage1 forward.
// All GEMMs use v_wmma_f32_16x16x32_f16 (wave32). LDS tiles are laid out so
// every WMMA fragment is fetched with ds_load_b128 (contiguous 16B runs).
// Global staging and the residual epilogue issue all b128/b32 loads before
// consuming them (multiple loads in flight, batched s_wait_loadcnt).
// ---------------------------------------------------------------------------

typedef __attribute__((ext_vector_type(16))) _Float16 v16h;
typedef __attribute__((ext_vector_type(8)))  _Float16 v8h;
typedef __attribute__((ext_vector_type(4)))  _Float16 v4h;
typedef __attribute__((ext_vector_type(8)))  float    v8f;

#define WMMA_F32_F16(a, b, c) \
  __builtin_amdgcn_wmma_f32_16x16x32_f16(false, (a), false, (b), (short)0, (c), false, false)

#define CAT16(lo, hi) \
  __builtin_shufflevector((lo), (hi), 0, 1, 2, 3, 4, 5, 6, 7, 8, 9, 10, 11, 12, 13, 14, 15)

// A fragment (16x32 f16): lanes<16 hold K 0..15, lanes>=16 hold K 16..31.
// One contiguous 16-half run -> two ds_load_b128.
__device__ __forceinline__ v16h load_a_frag(const _Float16* row, int lane) {
  const int kh = (lane & 16) ? 16 : 0;
  v8h lo = *(const v8h*)(row + kh);
  v8h hi = *(const v8h*)(row + kh + 8);
  return CAT16(lo, hi);
}
// B fragment (32x16 f16) from K-contiguous storage: lanes<16 hold K 0..15,
// lanes>=16 hold K 16..31. One contiguous 16-half run -> two ds_load_b128.
__device__ __forceinline__ v16h load_b_frag(const _Float16* colrow, int lane) {
  const int ko = (lane & 16) ? 16 : 0;
  v8h lo = *(const v8h*)(colrow + ko);
  v8h hi = *(const v8h*)(colrow + ko + 8);
  return CAT16(lo, hi);
}

__device__ __forceinline__ v4h cvt4(float4 f) {
  v4h h;
  h[0] = (_Float16)f.x; h[1] = (_Float16)f.y;
  h[2] = (_Float16)f.z; h[3] = (_Float16)f.w;
  return h;
}

// ---------------------------------------------------------------------------
// Patch embed: (8,3,224,224) -> tokens(25088) x 96, linear 48->96 + LayerNorm.
// ---------------------------------------------------------------------------
__global__ __launch_bounds__(128) void patch_embed_ln_kernel(
    const float* __restrict__ x, const float* __restrict__ pe_w,
    const float* __restrict__ pe_b, const float* __restrict__ g,
    const float* __restrict__ bb, float* __restrict__ out) {
  __shared__ float buf[4][48];
  const int tid = threadIdx.x, wave = tid >> 5, lane = tid & 31;
  const int tok0 = blockIdx.x * 4;
  for (int e = tid; e < 4 * 48; e += 128) {
    int tk = e / 48, f = e % 48;
    int t = tok0 + tk;
    int b = t / 3136, r = t % 3136, hi = r / 56, wi = r % 56;
    int i4 = f / 12, j4 = (f / 3) % 4, c = f % 3;
    buf[tk][f] = x[((b * 3 + c) * 224 + hi * 4 + i4) * 224 + wi * 4 + j4];
  }
  __syncthreads();
  const int t = tok0 + wave;
  float v[3];
  float s = 0.f, s2 = 0.f;
#pragma unroll
  for (int j = 0; j < 3; j++) {
    int co = lane + j * 32;
    float acc = pe_b[co];
#pragma unroll 8
    for (int f = 0; f < 48; f++) acc += buf[wave][f] * pe_w[f * 96 + co];
    v[j] = acc;
    s += acc;
    s2 += acc * acc;
  }
#pragma unroll
  for (int m = 16; m >= 1; m >>= 1) {
    s += __shfl_xor(s, m, 32);
    s2 += __shfl_xor(s2, m, 32);
  }
  float mean = s * (1.f / 96.f);
  float var = s2 * (1.f / 96.f) - mean * mean;
  float rstd = rsqrtf(var + 1e-5f);
#pragma unroll
  for (int j = 0; j < 3; j++) {
    int co = lane + j * 32;
    out[t * 96 + co] = (v[j] - mean) * rstd * g[co] + bb[co];
  }
}

// ---------------------------------------------------------------------------
// Generic LayerNorm over last dim C (32|C, C<=384). One wave per token.
// ---------------------------------------------------------------------------
__global__ __launch_bounds__(128) void ln_kernel(
    const float* __restrict__ x, float* __restrict__ y,
    const float* __restrict__ g, const float* __restrict__ b, int ntok, int C) {
  const int tid = threadIdx.x, wave = tid >> 5, lane = tid & 31;
  const int t = blockIdx.x * 4 + wave;
  if (t >= ntok) return;
  const int cpl = C >> 5;
  float v[12];
  float s = 0.f, s2 = 0.f;
  for (int j = 0; j < cpl; j++) {
    int ch = lane + j * 32;
    float vv = x[(long)t * C + ch];
    v[j] = vv;
    s += vv;
    s2 += vv * vv;
  }
#pragma unroll
  for (int m = 16; m >= 1; m >>= 1) {
    s += __shfl_xor(s, m, 32);
    s2 += __shfl_xor(s2, m, 32);
  }
  float inv = 1.f / (float)C;
  float mean = s * inv;
  float var = s2 * inv - mean * mean;
  float rstd = rsqrtf(var + 1e-5f);
  for (int j = 0; j < cpl; j++) {
    int ch = lane + j * 32;
    y[(long)t * C + ch] = (v[j] - mean) * rstd * g[ch] + b[ch];
  }
}

// ---------------------------------------------------------------------------
// Cyclic-shift + window partition gather: in (B,H,H,C) -> out (nw*49, C)
// ---------------------------------------------------------------------------
__global__ void gather_win_kernel(const float* __restrict__ in,
                                  float* __restrict__ out, int B, int H, int C,
                                  int shift) {
  long idx = (long)blockIdx.x * blockDim.x + threadIdx.x;
  long total = (long)B * H * H * C;
  if (idx >= total) return;
  int c = idx % C;
  long t = idx / C;
  int tok = t % 49;
  int win = t / 49;
  int nwg = H / 7, wpi = nwg * nwg;
  int b = win / wpi, wib = win % wpi;
  int wh = wib / nwg, ww = wib % nwg;
  int h = wh * 7 + tok / 7, w = ww * 7 + tok % 7;
  int hs = (h + shift) % H, ws = (w + shift) % H;
  out[idx] = in[((long)(b * H + hs) * H + ws) * C + c];
}

// Reverse: x(B,H,H,C) += windows(nw*49,C) mapped back through inverse shift.
__global__ void scatter_add_kernel(const float* __restrict__ win,
                                   float* __restrict__ x, int B, int H, int C,
                                   int shift) {
  long idx = (long)blockIdx.x * blockDim.x + threadIdx.x;
  long total = (long)B * H * H * C;
  if (idx >= total) return;
  int c = idx % C;
  long t = idx / C;
  int w = t % H;
  long t2 = t / H;
  int h = t2 % H;
  int b = t2 / H;
  int hs = (h - shift + H) % H, ws = (w - shift + H) % H;
  int nwg = H / 7;
  int wh = hs / 7, th = hs % 7, ww = ws / 7, tw = ws % 7;
  int winid = (b * nwg + wh) * nwg + ww;
  int tok = th * 7 + tw;
  x[idx] += win[((long)winid * 49 + tok) * C + c];
}

// ---------------------------------------------------------------------------
// WMMA GEMM: D[M,N] = act(A[M,K] @ W[K,N] + bias[N]) (+ residual R[M,N])
// block = 128 threads (4 waves) -> 64x32 output tile; wave = 16x32 (2 wmma/k).
// Requires M%64==0, N%32==0, K%32==0.
// ---------------------------------------------------------------------------
__global__ __launch_bounds__(128) void gemm_wmma_kernel(
    const float* __restrict__ A, const float* __restrict__ W,
    const float* __restrict__ bias, const float* __restrict__ R,
    float* __restrict__ D, int M, int N, int K, int act) {
  __shared__ _Float16 As[64][32];  // [row][k]
  __shared__ _Float16 Wt[32][32];  // [col][k] (transposed for b128 B-frags)
  const int tid = threadIdx.x, wave = tid >> 5, lane = tid & 31;
  const int rowBase = blockIdx.y * 64;
  const int colBase = blockIdx.x * 32;
  const int m0 = wave * 16;

  // Per-thread staging coordinates (invariant over k-steps).
  int ar[4], ac[4];
#pragma unroll
  for (int i = 0; i < 4; i++) {
    int e4 = tid + i * 128;            // 0..511
    ar[i] = e4 >> 3;                   // A row 0..63
    ac[i] = (e4 & 7) * 4;              // A col 0..28
  }
  int wr[2], wc[2];
#pragma unroll
  for (int i = 0; i < 2; i++) {
    int e4 = tid + i * 128;            // 0..255
    wr[i] = e4 >> 3;                   // W k-row 0..31
    wc[i] = (e4 & 7) * 4;              // W col 0..28
  }

  v8f acc0 = {}, acc1 = {};
  for (int k0 = 0; k0 < K; k0 += 32) {
    // Phase 1: issue ALL global b128 loads (6 outstanding per thread).
    float4 av[4], wv[2];
#pragma unroll
    for (int i = 0; i < 4; i++)
      av[i] = *(const float4*)&A[(long)(rowBase + ar[i]) * K + k0 + ac[i]];
#pragma unroll
    for (int i = 0; i < 2; i++)
      wv[i] = *(const float4*)&W[(long)(k0 + wr[i]) * N + colBase + wc[i]];
    if (k0 + 32 < K) {
      // CDNA5 speculative prefetch of next A tile (global_prefetch_b8)
      __builtin_prefetch(&A[(long)(rowBase + (tid >> 1)) * K + k0 + 32 + (tid & 1) * 16], 0, 0);
    }
    // Phase 2: convert + LDS stores.
#pragma unroll
    for (int i = 0; i < 4; i++) *(v4h*)&As[ar[i]][ac[i]] = cvt4(av[i]);
#pragma unroll
    for (int i = 0; i < 2; i++) {
      Wt[wc[i] + 0][wr[i]] = (_Float16)wv[i].x;
      Wt[wc[i] + 1][wr[i]] = (_Float16)wv[i].y;
      Wt[wc[i] + 2][wr[i]] = (_Float16)wv[i].z;
      Wt[wc[i] + 3][wr[i]] = (_Float16)wv[i].w;
    }
    __syncthreads();

    v16h af = load_a_frag(&As[m0 + (lane & 15)][0], lane);
    v16h b0 = load_b_frag(&Wt[(lane & 15)][0], lane);
    v16h b1 = load_b_frag(&Wt[16 + (lane & 15)][0], lane);
    acc0 = WMMA_F32_F16(af, b0, acc0);
    acc1 = WMMA_F32_F16(af, b1, acc1);
    __syncthreads();
  }

  // Epilogue: batch residual loads per 16-col tile, then convert + store.
  const int rowb = rowBase + m0 + ((lane & 16) ? 8 : 0);
#pragma unroll
  for (int nt = 0; nt < 2; nt++) {
    const v8f acc = nt ? acc1 : acc0;
    const int col = colBase + nt * 16 + (lane & 15);
    const float bv = bias ? bias[col] : 0.f;
    float rv[8];
    if (R) {
#pragma unroll
      for (int r = 0; r < 8; r++) rv[r] = R[(long)(rowb + r) * N + col];
    } else {
#pragma unroll
      for (int r = 0; r < 8; r++) rv[r] = 0.f;
    }
#pragma unroll
    for (int r = 0; r < 8; r++) {
      float v = acc[r] + bv;
      if (act == 1) v = 0.5f * v * (1.f + erff(v * 0.70710678118654752f));
      D[(long)(rowb + r) * N + col] = v + rv[r];
    }
  }
}

// ---------------------------------------------------------------------------
// Window attention, one block per (window, head). N=49 padded to 64, d=32.
// ---------------------------------------------------------------------------
__global__ __launch_bounds__(128) void attn_wmma_kernel(
    const float* __restrict__ qkv, float* __restrict__ o,
    const float* __restrict__ rel, int C, int heads, int nwW, int shifted,
    int H) {
  __shared__ _Float16 Qs[64][32];   // [tok][d]
  __shared__ _Float16 Ks[64][32];   // [tok][d]  (B-frag reads are k-contiguous)
  __shared__ _Float16 Vt[32][64];   // [d][tok]  (transposed for P@V B-frags)
  __shared__ float    S[64][64];    // logits
  __shared__ _Float16 Ph[64][64];   // probabilities (A operand of P@V)

  const int tid = threadIdx.x, wave = tid >> 5, lane = tid & 31;
  const int win = blockIdx.x / heads;
  const int head = blockIdx.x % heads;
  const int wpi = nwW * nwW;
  const int wib = win % wpi;
  const int wh = wib / nwW, ww = wib % nwW;
  const int C3 = 3 * C;
  const float qs = 0.17677669529663687f;  // 32^-0.5

  // Stage Q,K,V (64x32 each, rows >=49 zero). Issue all 12 b128 loads first.
  float4 qv[4], kv[4], vv[4];
  int tn[4], tc[4];
#pragma unroll
  for (int i = 0; i < 4; i++) {
    int e4 = tid + i * 128;            // 0..511
    tn[i] = e4 >> 3;                   // token row 0..63
    tc[i] = (e4 & 7) * 4;              // d col
    if (tn[i] < 49) {
      long row = ((long)win * 49 + tn[i]) * C3 + head * 32 + tc[i];
      qv[i] = *(const float4*)&qkv[row];
      kv[i] = *(const float4*)&qkv[row + C];
      vv[i] = *(const float4*)&qkv[row + 2 * C];
    } else {
      qv[i] = make_float4(0.f, 0.f, 0.f, 0.f);
      kv[i] = qv[i];
      vv[i] = qv[i];
    }
  }
#pragma unroll
  for (int i = 0; i < 4; i++) {
    float4 q = qv[i];
    q.x *= qs; q.y *= qs; q.z *= qs; q.w *= qs;
    *(v4h*)&Qs[tn[i]][tc[i]] = cvt4(q);
    *(v4h*)&Ks[tn[i]][tc[i]] = cvt4(kv[i]);
    Vt[tc[i] + 0][tn[i]] = (_Float16)vv[i].x;
    Vt[tc[i] + 1][tn[i]] = (_Float16)vv[i].y;
    Vt[tc[i] + 2][tn[i]] = (_Float16)vv[i].z;
    Vt[tc[i] + 3][tn[i]] = (_Float16)vv[i].w;
  }
  __syncthreads();

  // S = Q @ K^T (64x64, K=32); wave handles rows [16*wave, 16*wave+16)
  const int m0 = wave * 16;
  {
    v16h af = load_a_frag(&Qs[m0 + (lane & 15)][0], lane);
#pragma unroll
    for (int j = 0; j < 4; j++) {
      v16h bf = load_b_frag(&Ks[j * 16 + (lane & 15)][0], lane);
      v8f c = {};
      c = WMMA_F32_F16(af, bf, c);
      const int rr = m0 + ((lane & 16) ? 8 : 0);
#pragma unroll
      for (int r = 0; r < 8; r++) S[rr + r][j * 16 + (lane & 15)] = c[r];
    }
  }
  __syncthreads();

  // bias + shift-mask + softmax over 49 valid cols -> Ph (f16)
  if (tid < 64) {
    int n = tid;
    if (n < 49) {
      int ni = n / 7, nj = n % 7;
      int rgn = 0;
      if (shifted) {
        int hn = wh * 7 + ni, wn = ww * 7 + nj;
        int rr = (hn < H - 7) ? 0 : (hn < H - 3 ? 1 : 2);
        int rc = (wn < H - 7) ? 0 : (wn < H - 3 ? 1 : 2);
        rgn = rr * 3 + rc;
      }
      float mx = -1e30f;
      for (int m = 0; m < 49; m++) {
        int mi = m / 7, mj = m % 7;
        float a = S[n][m] + rel[((ni - mi + 6) * 13 + (nj - mj + 6)) * heads + head];
        if (shifted) {
          int hm = wh * 7 + mi, wm = ww * 7 + mj;
          int rr = (hm < H - 7) ? 0 : (hm < H - 3 ? 1 : 2);
          int rc = (wm < H - 7) ? 0 : (wm < H - 3 ? 1 : 2);
          if (rr * 3 + rc != rgn) a += -100.f;
        }
        S[n][m] = a;
        mx = fmaxf(mx, a);
      }
      float sum = 0.f;
      for (int m = 0; m < 49; m++) {
        float e = __expf(S[n][m] - mx);
        S[n][m] = e;
        sum += e;
      }
      float inv = 1.f / sum;
      for (int m = 0; m < 49; m++) Ph[n][m] = (_Float16)(S[n][m] * inv);
      for (int m = 49; m < 64; m++) Ph[n][m] = (_Float16)0.f;
    } else {
      for (int m = 0; m < 64; m++) Ph[n][m] = (_Float16)0.f;
    }
  }
  __syncthreads();

  // O = P @ V (64x32, K=64 -> two K steps of 32)
#pragma unroll
  for (int jn = 0; jn < 2; jn++) {
    const int n0 = jn * 16;
    v8f c = {};
#pragma unroll
    for (int ks = 0; ks < 2; ks++) {
      const int k0 = ks * 32;
      v16h af = load_a_frag(&Ph[m0 + (lane & 15)][k0], lane);
      v16h bf = load_b_frag(&Vt[n0 + (lane & 15)][k0], lane);
      c = WMMA_F32_F16(af, bf, c);
    }
    const int rr = m0 + ((lane & 16) ? 8 : 0);
#pragma unroll
    for (int r = 0; r < 8; r++) {
      int row = rr + r;
      if (row < 49)
        o[((long)win * 49 + row) * C + head * 32 + n0 + (lane & 15)] = c[r];
    }
  }
}

// ---------------------------------------------------------------------------
// Patch-merge gather: x(B,56,56,96) -> xm(B*28*28, 384), quadrant concat.
// ---------------------------------------------------------------------------
__global__ void gather_merge_kernel(const float* __restrict__ x,
                                    float* __restrict__ xm, int B) {
  long idx = (long)blockIdx.x * blockDim.x + threadIdx.x;
  long total = (long)B * 28 * 28 * 384;
  if (idx >= total) return;
  int f = idx % 384;
  long t = idx / 384;
  int j = t % 28;
  long t2 = t / 28;
  int i = t2 % 28;
  int b = t2 / 28;
  int q = f / 96, c = f % 96;
  int dh = q & 1, dw = (q >> 1) & 1;
  xm[idx] = x[((long)(b * 56 + 2 * i + dh) * 56 + (2 * j + dw)) * 96 + c];
}

__global__ void copy_kernel(const float* __restrict__ in, float* __restrict__ out,
                            long n) {
  long i = (long)blockIdx.x * blockDim.x + threadIdx.x;
  if (i < n) out[i] = in[i];
}

// ---------------------------------------------------------------------------
// Host side
// ---------------------------------------------------------------------------
struct BlkP {
  const float *fc1_b, *fc1_w, *fc2_b, *fc2_w, *ln1_b, *ln1_g, *ln2_b, *ln2_g;
  const float *proj_b, *proj_w, *qkv_b, *qkv_w, *rel;
};

static BlkP get_blk(void* const* d_in, int base) {
  BlkP p;
  p.fc1_b = (const float*)d_in[base + 0];
  p.fc1_w = (const float*)d_in[base + 1];
  p.fc2_b = (const float*)d_in[base + 2];
  p.fc2_w = (const float*)d_in[base + 3];
  p.ln1_b = (const float*)d_in[base + 4];
  p.ln1_g = (const float*)d_in[base + 5];
  p.ln2_b = (const float*)d_in[base + 6];
  p.ln2_g = (const float*)d_in[base + 7];
  p.proj_b = (const float*)d_in[base + 8];
  p.proj_w = (const float*)d_in[base + 9];
  p.qkv_b = (const float*)d_in[base + 10];
  p.qkv_w = (const float*)d_in[base + 11];
  p.rel = (const float*)d_in[base + 12];
  return p;
}

static void launch_gemm(hipStream_t s, const float* A, const float* W,
                        const float* bias, const float* R, float* D, int M,
                        int N, int K, int act) {
  dim3 g(N / 32, M / 64);
  gemm_wmma_kernel<<<g, 128, 0, s>>>(A, W, bias, R, D, M, N, K, act);
}

static void run_block(hipStream_t s, const BlkP& p, float* x, float* t,
                      float* w, float* big, int M, int C, int heads, int H,
                      int shift) {
  const int B = 8;
  const long elems = (long)M * C;
  const int eb = (int)((elems + 255) / 256);
  const int nwW = H / 7;
  const int nw = B * nwW * nwW;

  ln_kernel<<<(M + 3) / 4, 128, 0, s>>>(x, t, p.ln1_g, p.ln1_b, M, C);
  gather_win_kernel<<<eb, 256, 0, s>>>(t, w, B, H, C, shift);
  launch_gemm(s, w, p.qkv_w, p.qkv_b, nullptr, big, M, 3 * C, C, 0);
  attn_wmma_kernel<<<nw * heads, 128, 0, s>>>(big, w, p.rel, C, heads, nwW,
                                              shift > 0 ? 1 : 0, H);
  launch_gemm(s, w, p.proj_w, p.proj_b, nullptr, t, M, C, C, 0);
  scatter_add_kernel<<<eb, 256, 0, s>>>(t, x, B, H, C, shift);
  ln_kernel<<<(M + 3) / 4, 128, 0, s>>>(x, t, p.ln2_g, p.ln2_b, M, C);
  launch_gemm(s, t, p.fc1_w, p.fc1_b, nullptr, big, M, 4 * C, C, 1);
  launch_gemm(s, big, p.fc2_w, p.fc2_b, x, x, M, C, 4 * C, 0);
}

extern "C" void kernel_launch(void* const* d_in, const int* in_sizes, int n_in,
                              void* d_out, int out_size, void* d_ws,
                              size_t ws_size, hipStream_t stream) {
  const float* x_in = (const float*)d_in[0];
  const float* m_ln_b = (const float*)d_in[1];
  const float* m_ln_g = (const float*)d_in[2];
  const float* m_red_w = (const float*)d_in[3];
  const float* pe_b = (const float*)d_in[4];
  const float* pe_ln_b = (const float*)d_in[5];
  const float* pe_ln_g = (const float*)d_in[6];
  const float* pe_w = (const float*)d_in[7];
  BlkP s00 = get_blk(d_in, 8);
  BlkP s01 = get_blk(d_in, 21);
  BlkP s10 = get_blk(d_in, 34);
  BlkP s11 = get_blk(d_in, 47);

  // Workspace layout (floats). Needs ~67.5 MB.
  float* ws = (float*)d_ws;
  const long A0 = 25088L * 96;  // 2,408,448
  float* R1 = ws;               // activations x
  float* R2 = ws + A0;          // ln / temp
  float* R3 = ws + 2 * A0;      // windowed / attn-out
  float* R4 = ws + 3 * A0;      // qkv / mlp hidden (9,633,792 floats)

  const int M0 = 25088, C0 = 96, H0 = 56;
  const int M1 = 6272, C1 = 192, H1 = 28;

  // Stage 0
  patch_embed_ln_kernel<<<M0 / 4, 128, 0, stream>>>(x_in, pe_w, pe_b, pe_ln_g,
                                                    pe_ln_b, R1);
  run_block(stream, s00, R1, R2, R3, R4, M0, C0, 3, H0, 0);
  run_block(stream, s01, R1, R2, R3, R4, M0, C0, 3, H0, 3);

  // Patch merge: 56x56x96 -> 28x28x384 -> LN -> @red_w -> 28x28x192
  {
    long total = (long)M1 * 384;
    gather_merge_kernel<<<(int)((total + 255) / 256), 256, 0, stream>>>(R1, R4, 8);
    ln_kernel<<<(M1 + 3) / 4, 128, 0, stream>>>(R4, R2, m_ln_g, m_ln_b, M1, 384);
    launch_gemm(stream, R2, m_red_w, nullptr, nullptr, R1, M1, 192, 384, 0);
  }

  // Stage 1
  run_block(stream, s10, R1, R2, R3, R4, M1, C1, 6, H1, 0);
  run_block(stream, s11, R1, R2, R3, R4, M1, C1, 6, H1, 3);

  // Output: (8,28,28,192) = 1,204,224 floats
  long n = (long)out_size;
  copy_kernel<<<(int)((n + 255) / 256), 256, 0, stream>>>(R1, (float*)d_out, n);
}
